// Attention_37778532336009
// MI455X (gfx1250) — matrix operand order repaired and checked by reading, not compiled, not run on hardware
//
#include <hip/hip_runtime.h>
#include <hip/hip_bf16.h>

typedef __attribute__((ext_vector_type(16))) _Float16 v16h;
typedef __attribute__((ext_vector_type(8)))  _Float16 v8h;
typedef __attribute__((ext_vector_type(8)))  float    v8f;

#define B_  2
#define S_  2048
#define D_  1024
#define H_  16
#define DH_ 64

__device__ __forceinline__ v8f wmma_f16f32(v16h a, v16h b, v8f c) {
  return __builtin_amdgcn_wmma_f32_16x16x32_f16(false, a, false, b, (short)0, c,
                                                false, false);
}

// A fragment: logical A[m][k], row-major source with leading dim ld.
// Lane m = lane&15, half = lane>>4. elems 0..15 -> k = half*16+i.
__device__ __forceinline__ v16h load_a_frag(const _Float16* base, int ld, int m0, int k0) {
  const int lane = threadIdx.x & 31;
  const int m = lane & 15, half = lane >> 4;
  const _Float16* p = base + (size_t)(m0 + m) * ld + (k0 + half * 16);
  v8h lo = *(const v8h*)p;
  v8h hi = *(const v8h*)(p + 8);
  v16h a;
#pragma unroll
  for (int i = 0; i < 8; ++i) { a[i] = lo[i]; a[i + 8] = hi[i]; }
  return a;
}

// B fragment of logical B[k][n]; source stored transposed BT[n][k] (row-major, ld=ldk)
// so the per-lane 16 consecutive K values are one contiguous 32B load.
// Lane n = lane&15; half = lane>>4 selects K group (0..15 / 16..31).
__device__ __forceinline__ v16h load_b_fragT(const _Float16* baseT, int ldk, int n0, int k0) {
  const int lane = threadIdx.x & 31;
  const int n = lane & 15, half = lane >> 4;
  const _Float16* p = baseT + (size_t)(n0 + n) * ldk + (k0 + half * 16);
  return *(const v16h*)p;
}

__global__ void cvt_f32_to_f16(const float* __restrict__ src, _Float16* __restrict__ dst, int n) {
  for (int i = blockIdx.x * blockDim.x + threadIdx.x; i < n; i += gridDim.x * blockDim.x)
    dst[i] = (_Float16)src[i];
}

// W [H][D][DH] f32 -> Wt [H][DH][D] f16   (so B-fragments over K=D are contiguous)
__global__ void pack_w_hde(const float* __restrict__ W, _Float16* __restrict__ Wt) {
  int idx = blockIdx.x * blockDim.x + threadIdx.x;   // exactly H_*D_*DH_ threads
  int h = idx / (D_ * DH_);
  int r = idx - h * (D_ * DH_);
  int d = r / DH_;
  int e = r - d * DH_;
  Wt[((size_t)h * DH_ + e) * D_ + d] = (_Float16)W[idx];
}

// W_O [H][DH][D] f32 -> WoT [D][H*DH] f16
__global__ void pack_wo(const float* __restrict__ W, _Float16* __restrict__ Wt) {
  int idx = blockIdx.x * blockDim.x + threadIdx.x;   // exactly H_*DH_*D_ threads
  int h = idx / (DH_ * D_);
  int r = idx - h * (DH_ * D_);
  int e = r / D_;
  int d = r - e * D_;
  Wt[(size_t)d * (H_ * DH_) + h * DH_ + e] = (_Float16)W[idx];
}

// Fused Q/K/V projection. Block = 256 thr = 8 waves, each wave: 16 rows x 64 cols,
// K-loop over D in steps of 32 -> 12 WMMAs/iter.
// Q,K stored [b][h][s][dh]; V stored transposed [b][h][dh][s].
__global__ void __launch_bounds__(256)
qkv_proj(const _Float16* __restrict__ xh,
         const _Float16* __restrict__ WqT, const _Float16* __restrict__ WkT,
         const _Float16* __restrict__ WvT,
         const float* __restrict__ bQ, const float* __restrict__ bK,
         const float* __restrict__ bV,
         _Float16* __restrict__ Qd, _Float16* __restrict__ Kd, _Float16* __restrict__ Vt) {
  const int sblocks = S_ / 128;
  int bid = blockIdx.x;
  int sblk = bid % sblocks; bid /= sblocks;
  int h = bid % H_;
  int b = bid / H_;
  int w = threadIdx.x >> 5;
  int s0 = sblk * 128 + w * 16;

  const _Float16* xb = xh + (size_t)b * S_ * D_;
  const _Float16* wq = WqT + (size_t)h * DH_ * D_;
  const _Float16* wk = WkT + (size_t)h * DH_ * D_;
  const _Float16* wv = WvT + (size_t)h * DH_ * D_;

  v8f zero = {};
  v8f aq[4], ak[4], av[4];
#pragma unroll
  for (int j = 0; j < 4; ++j) { aq[j] = zero; ak[j] = zero; av[j] = zero; }

  for (int kd = 0; kd < D_; kd += 32) {
    v16h af = load_a_frag(xb, D_, s0, kd);
#pragma unroll
    for (int j = 0; j < 4; ++j) {
      aq[j] = wmma_f16f32(af, load_b_fragT(wq, D_, j * 16, kd), aq[j]);
      ak[j] = wmma_f16f32(af, load_b_fragT(wk, D_, j * 16, kd), ak[j]);
      av[j] = wmma_f16f32(af, load_b_fragT(wv, D_, j * 16, kd), av[j]);
    }
  }

  const int lane = threadIdx.x & 31;
  const int n = lane & 15, half = lane >> 4;
  size_t ho = (size_t)b * H_ + h;
  _Float16* Qb = Qd + ho * S_ * DH_;
  _Float16* Kb = Kd + ho * S_ * DH_;
  _Float16* Vb = Vt + ho * DH_ * S_;
#pragma unroll
  for (int j = 0; j < 4; ++j) {
    int col = j * 16 + n;
    float qb = bQ[h * DH_ + col];
    float kb = bK[h * DH_ + col];
    float vb = bV[h * DH_ + col];
#pragma unroll
    for (int i = 0; i < 8; ++i) {
      int row = s0 + half * 8 + i;
      Qb[(size_t)row * DH_ + col] = (_Float16)(aq[j][i] + qb);
      Kb[(size_t)row * DH_ + col] = (_Float16)(ak[j][i] + kb);
      Vb[(size_t)col * S_ + row] = (_Float16)(av[j][i] + vb);
    }
  }
}

// Flash attention. Each wave owns 16 query rows; per-wave private LDS tile used to
// convert P from C-layout (lane=key) to A-layout (lane=row) between QK^T and PV.
__global__ void __launch_bounds__(256)
flash_attn(const _Float16* __restrict__ Qd, const _Float16* __restrict__ Kd,
           const _Float16* __restrict__ Vt, _Float16* __restrict__ Z) {
  __shared__ __align__(16) _Float16 ldsP[8 * 16 * 32];   // 8KB: per-wave 16x32 P tile
  const int sblocks = S_ / 128;
  int bid = blockIdx.x;
  int sblk = bid % sblocks; bid /= sblocks;
  int h = bid % H_;
  int b = bid / H_;
  int w = threadIdx.x >> 5;
  int q0 = sblk * 128 + w * 16;
  const int lane = threadIdx.x & 31;
  const int n = lane & 15, half = lane >> 4;

  size_t ho = (size_t)b * H_ + h;
  const _Float16* Qb = Qd + ho * S_ * DH_;
  const _Float16* Kb = Kd + ho * S_ * DH_;
  const _Float16* Vb = Vt + ho * DH_ * S_;

  v16h qf0 = load_a_frag(Qb, DH_, q0, 0);
  v16h qf1 = load_a_frag(Qb, DH_, q0, 32);

  v8f zero = {};
  v8f zacc[4];
#pragma unroll
  for (int t = 0; t < 4; ++t) zacc[t] = zero;
  float mrun[8], lrun[8];
#pragma unroll
  for (int j = 0; j < 8; ++j) { mrun[j] = -3.0e38f; lrun[j] = 0.0f; }

  _Float16* pbuf = &ldsP[w * 512];
  int nkb = q0 / 32 + 1;                       // causal: keys <= q0+15
  for (int kb = 0; kb < nkb; ++kb) {
    int key0 = kb * 32;
    // S tile 16x32: B-frags read K row-contiguously (K^T with lane = key row)
    v8f s0 = zero, s1 = zero;
    s0 = wmma_f16f32(qf0, load_b_fragT(Kb, DH_, key0, 0), s0);
    s0 = wmma_f16f32(qf1, load_b_fragT(Kb, DH_, key0, 32), s0);
    s1 = wmma_f16f32(qf0, load_b_fragT(Kb, DH_, key0 + 16, 0), s1);
    s1 = wmma_f16f32(qf1, load_b_fragT(Kb, DH_, key0 + 16, 32), s1);

#pragma unroll
    for (int j = 0; j < 8; ++j) {
      int row = q0 + half * 8 + j;
      float v0 = (key0 + n <= row)      ? s0[j] * 0.125f : -1.0e5f;
      float v1 = (key0 + 16 + n <= row) ? s1[j] * 0.125f : -1.0e5f;
      float bm = fmaxf(v0, v1);                     // row max across 16-lane group
      bm = fmaxf(bm, __shfl_xor(bm, 1));
      bm = fmaxf(bm, __shfl_xor(bm, 2));
      bm = fmaxf(bm, __shfl_xor(bm, 4));
      bm = fmaxf(bm, __shfl_xor(bm, 8));
      float mnew  = fmaxf(mrun[j], bm);
      float alpha = __expf(mrun[j] - mnew);
      float p0 = __expf(v0 - mnew);
      float p1 = __expf(v1 - mnew);
      float rs = p0 + p1;
      rs += __shfl_xor(rs, 1);
      rs += __shfl_xor(rs, 2);
      rs += __shfl_xor(rs, 4);
      rs += __shfl_xor(rs, 8);
      lrun[j] = lrun[j] * alpha + rs;
      mrun[j] = mnew;
#pragma unroll
      for (int t = 0; t < 4; ++t) zacc[t][j] *= alpha;
      int rl = half * 8 + j;
      pbuf[rl * 32 + n]      = (_Float16)p0;        // stage P row-major in LDS
      pbuf[rl * 32 + 16 + n] = (_Float16)p1;
    }
    asm volatile("s_wait_dscnt 0" ::: "memory");    // within-wave LDS RAW
    // reload P as A-fragment (lane = query row) and do PV against transposed V
    {
      const _Float16* pr = pbuf + (lane & 15) * 32 + half * 16;
      v8h lo = *(const v8h*)pr;
      v8h hi = *(const v8h*)(pr + 8);
      v16h pf;
#pragma unroll
      for (int i = 0; i < 8; ++i) { pf[i] = lo[i]; pf[i + 8] = hi[i]; }
#pragma unroll
      for (int t = 0; t < 4; ++t)
        zacc[t] = wmma_f16f32(pf, load_b_fragT(Vb, S_, t * 16, key0), zacc[t]);
    }
  }

  // normalize and store Z as [b][s][h*DH] so the out-proj A-fragments are contiguous
#pragma unroll
  for (int t = 0; t < 4; ++t) {
#pragma unroll
    for (int j = 0; j < 8; ++j) {
      int row = q0 + half * 8 + j;
      float zv = zacc[t][j] / lrun[j];
      Z[((size_t)b * S_ + row) * (H_ * DH_) + h * DH_ + t * 16 + n] = (_Float16)zv;
    }
  }
}

// attn_out = Z[4096x1024] @ W_O^T-packed[1024x1024] + b_O, fp32 out.
__global__ void __launch_bounds__(256)
out_proj(const _Float16* __restrict__ Z, const _Float16* __restrict__ WoT,
         const float* __restrict__ bO, float* __restrict__ out) {
  const int cblocks = D_ / 64;   // 16
  int bid = blockIdx.x;
  int cblk = bid % cblocks;
  int rblk = bid / cblocks;
  int w = threadIdx.x >> 5;
  int r0 = rblk * 128 + w * 16;
  int c0 = cblk * 64;

  v8f zero = {};
  v8f acc[4];
#pragma unroll
  for (int j = 0; j < 4; ++j) acc[j] = zero;

  for (int kc = 0; kc < H_ * DH_; kc += 32) {
    v16h af = load_a_frag(Z, H_ * DH_, r0, kc);
#pragma unroll
    for (int j = 0; j < 4; ++j)
      acc[j] = wmma_f16f32(af, load_b_fragT(WoT, H_ * DH_, c0 + j * 16, kc), acc[j]);
  }

  const int lane = threadIdx.x & 31;
  const int n = lane & 15, half = lane >> 4;
#pragma unroll
  for (int j = 0; j < 4; ++j) {
    int col = c0 + j * 16 + n;
    float bo = bO[col];
#pragma unroll
    for (int i = 0; i < 8; ++i)
      out[(size_t)(r0 + half * 8 + i) * D_ + col] = acc[j][i] + bo;
  }
}

extern "C" void kernel_launch(void* const* d_in, const int* in_sizes, int n_in,
                              void* d_out, int out_size, void* d_ws, size_t ws_size,
                              hipStream_t stream) {
  (void)in_sizes; (void)n_in; (void)out_size; (void)ws_size;
  const float* x  = (const float*)d_in[0];
  const float* WQ = (const float*)d_in[1];
  const float* WK = (const float*)d_in[2];
  const float* WV = (const float*)d_in[3];
  const float* WO = (const float*)d_in[4];
  const float* bQ = (const float*)d_in[5];
  const float* bK = (const float*)d_in[6];
  const float* bV = (const float*)d_in[7];
  const float* bO = (const float*)d_in[8];
  float* out = (float*)d_out;

  char* ws = (char*)d_ws;
  size_t off = 0;
  auto wsalloc = [&](size_t bytes) -> void* {
    void* p = ws + off;
    off += (bytes + 255) & ~(size_t)255;
    return p;
  };
  const size_t nX = (size_t)B_ * S_ * D_;          // 4,194,304
  const size_t nW = (size_t)H_ * DH_ * D_;         // 1,048,576
  const size_t nQ = (size_t)B_ * H_ * S_ * DH_;    // 4,194,304

  _Float16* xh  = (_Float16*)wsalloc(nX * 2);
  _Float16* WqT = (_Float16*)wsalloc(nW * 2);
  _Float16* WkT = (_Float16*)wsalloc(nW * 2);
  _Float16* WvT = (_Float16*)wsalloc(nW * 2);
  _Float16* WoT = (_Float16*)wsalloc(nW * 2);
  _Float16* Qd  = (_Float16*)wsalloc(nQ * 2);
  _Float16* Kd  = (_Float16*)wsalloc(nQ * 2);
  _Float16* Vt  = (_Float16*)wsalloc(nQ * 2);
  _Float16* Zb  = (_Float16*)wsalloc(nX * 2);
  // total ~50.3 MB of d_ws

  cvt_f32_to_f16<<<2048, 256, 0, stream>>>(x, xh, (int)nX);
  pack_w_hde<<<(int)(nW / 256), 256, 0, stream>>>(WQ, WqT);
  pack_w_hde<<<(int)(nW / 256), 256, 0, stream>>>(WK, WkT);
  pack_w_hde<<<(int)(nW / 256), 256, 0, stream>>>(WV, WvT);
  pack_wo  <<<(int)(nW / 256), 256, 0, stream>>>(WO, WoT);

  qkv_proj<<<B_ * H_ * (S_ / 128), 256, 0, stream>>>(xh, WqT, WkT, WvT,
                                                     bQ, bK, bV, Qd, Kd, Vt);
  flash_attn<<<B_ * H_ * (S_ / 128), 256, 0, stream>>>(Qd, Kd, Vt, Zb);
  out_proj<<<(B_ * S_ / 128) * (D_ / 64), 256, 0, stream>>>(Zb, WoT, bO, out);
}
